// ConvS5SSM_56083682951460
// MI455X (gfx1250) — matrix operand order repaired and checked by reading, not compiled, not run on hardware
//
#include <hip/hip_runtime.h>
#include <hip/hip_bf16.h>
#include <math.h>

typedef __attribute__((ext_vector_type(16))) _Float16 v16h;
typedef __attribute__((ext_vector_type(8)))  float    v8f;

#define LSEQ 16
#define BB   8
#define HH   32
#define WW   32
#define UU   128
#define PP   128
#define LB   (LSEQ*BB)                      // 128
#define NPIX ((size_t)BB*HH*WW*PP)          // scan width: 1,048,576
#define PLANEH ((size_t)LB*HH*WW*PP)        // halves per Bu/xs plane: 16,777,216

// ---- workspace layout (byte offsets, all 256B-aligned) ----
#define O_DISC 0u                            // 512 floats (A.re A.im coef.re coef.im)
#define O_BW   ((size_t)2048)                // packed B weights: 576 tiles * 1024 B
#define O_CW   (O_BW + 589824)               // packed C weights: 576 tiles * 1024 B
#define O_UH   (O_CW + 589824)               // u as f16: 16,777,216 halves
#define O_BURE (O_UH + 2*PLANEH)             // Bu/xs real plane, f16
#define O_BUIM (O_BURE + 2*PLANEH)           // Bu/xs imag plane, f16

// ---------------------------------------------------------------------------
// Async global->LDS copy (CDNA5 ASYNCcnt path), with safe fallback
// ---------------------------------------------------------------------------
#if defined(__has_builtin)
#if __has_builtin(__builtin_amdgcn_global_load_async_to_lds_b128) && \
    __has_builtin(__builtin_amdgcn_s_wait_asynccnt)
#define ASYNC_LDS 1
#endif
#endif

typedef int v4i_gcc __attribute__((vector_size(16)));
typedef __attribute__((address_space(1))) v4i_gcc* as1_v4i;
typedef __attribute__((address_space(3))) v4i_gcc* as3_v4i;

__device__ __forceinline__ void copy16_g2lds(const void* g, void* l) {
#ifdef ASYNC_LDS
  void* gnc = const_cast<void*>(g);
  __builtin_amdgcn_global_load_async_to_lds_b128(
      (as1_v4i)gnc, (as3_v4i)l, 0, 0);
#else
  *(uint4*)l = *(const uint4*)g;
#endif
}

__device__ __forceinline__ void wait_async_copies() {
#ifdef ASYNC_LDS
  __builtin_amdgcn_s_wait_asynccnt(0);
#endif
}

// ---------------------------------------------------------------------------
// WMMA helpers (gfx1250, wave32)
// ---------------------------------------------------------------------------
__device__ __forceinline__ v8f wmma16x16x32(v16h a, v16h b, v8f c) {
  return __builtin_amdgcn_wmma_f32_16x16x32_f16(false, a, false, b, (short)0, c, false, false);
}

// A-fragment (16x32 f16) from an LDS row pointer, contiguous k order:
// lane m = l&15, half = l>>4; VGPR i holds K = 2i + 16*half, +1
__device__ __forceinline__ v16h make_afrag(const _Float16* rowp, int half) {
  union { unsigned int u[8]; v16h v; } x;
  const unsigned int* rp = (const unsigned int*)rowp;
  #pragma unroll
  for (int i = 0; i < 8; ++i) {
    int kb = 2*i + 16*half;   // always even
    x.u[i] = rp[kb >> 1];
  }
  return x.v;
}

// B-fragment from pre-packed global tile: 256 dwords, lane-major (lane*8 + i)
__device__ __forceinline__ v16h load_bfrag(const unsigned int* tile) {
  union { unsigned int u[8]; v16h v; } x;
  const unsigned int* p = tile + (threadIdx.x & 31) * 8;
  #pragma unroll
  for (int i = 0; i < 8; ++i) x.u[i] = p[i];
  return x.v;
}

// ---------------------------------------------------------------------------
// Kernel 0: ZOH discretization
// ---------------------------------------------------------------------------
__global__ void disc_kernel(const float* __restrict__ Lre,
                            const float* __restrict__ Lim,
                            const float* __restrict__ logstep,
                            float* __restrict__ ws) {
  int p = threadIdx.x;
  if (p >= PP) return;
  float lr = fminf(Lre[p], -1e-4f);
  float li = Lim[p];
  float s  = expf(logstep[p]);
  float er = expf(lr * s);
  float Are = er * cosf(li * s);
  float Aim = er * sinf(li * s);
  float den = lr * lr + li * li;
  float cr = ((Are - 1.0f) * lr + Aim * li) / den;
  float ci = (Aim * lr - (Are - 1.0f) * li) / den;
  ws[p] = Are; ws[128 + p] = Aim; ws[256 + p] = cr; ws[384 + p] = ci;
}

// ---------------------------------------------------------------------------
// Kernel 1: f32 -> f16 convert of input_sequence
// ---------------------------------------------------------------------------
__global__ __launch_bounds__(256)
void cvt_kernel(const float* __restrict__ u, _Float16* __restrict__ uh) {
  size_t i = ((size_t)blockIdx.x * 256 + threadIdx.x) * 2;
  uh[i]     = (_Float16)u[i];
  uh[i + 1] = (_Float16)u[i + 1];
}

// ---------------------------------------------------------------------------
// Kernel 2: pack B_bar = (B.re + i B.im) * coef into f16 B-fragment tiles.
// tile id = ((t*4 + kc)*8 + nTile)*2 + reim ; 256 dwords per tile (lane*8+i)
// ---------------------------------------------------------------------------
__global__ void packB_kernel(const float* __restrict__ Bk,   // (3,3,U,P,2)
                             const float* __restrict__ ws,
                             _Float16* __restrict__ Bw) {
  int tile = blockIdx.x;            // 0..575
  int q = threadIdx.x;              // dword within tile
  int lane = q >> 3, i = q & 7;
  int reim = tile & 1;
  int nt   = (tile >> 1) & 7;
  int kc   = (tile >> 4) & 3;
  int t    = tile >> 6;
  int p_ch = nt * 16 + (lane & 15);
  float cr = ws[256 + p_ch], ci = ws[384 + p_ch];
  #pragma unroll
  for (int s = 0; s < 2; ++s) {
    int u_ch = kc * 32 + 16 * (lane >> 4) + 2 * i + s;
    const float* src = Bk + (((size_t)t * UU + u_ch) * PP + p_ch) * 2;
    float br = src[0], bi = src[1];
    float v  = (reim == 0) ? (br * cr - bi * ci) : (br * ci + bi * cr);
    Bw[(size_t)tile * 512 + q * 2 + s] = (_Float16)v;
  }
}

// ---------------------------------------------------------------------------
// Kernel 3: pack [2*C.re ; -2*C.im] into f16 fragment tiles.
// tile id = (((rp*9 + t)*4 + kc)*8 + nTile)
// ---------------------------------------------------------------------------
__global__ void packC_kernel(const float* __restrict__ Ck,   // (3,3,P,U,2)
                             _Float16* __restrict__ Cw) {
  int tile = blockIdx.x;
  int q = threadIdx.x;
  int lane = q >> 3, i = q & 7;
  int nt = tile & 7;
  int kc = (tile >> 3) & 3;
  int rem = tile >> 5;              // rp*9 + t
  int t  = rem % 9;
  int rp = rem / 9;
  int u_ch = nt * 16 + (lane & 15);
  #pragma unroll
  for (int s = 0; s < 2; ++s) {
    int p_ch = kc * 32 + 16 * (lane >> 4) + 2 * i + s;
    const float* src = Ck + (((size_t)t * PP + p_ch) * UU + u_ch) * 2;
    float v = (rp == 0) ? (2.0f * src[0]) : (-2.0f * src[1]);
    Cw[(size_t)tile * 512 + q * 2 + s] = (_Float16)v;
  }
}

// ---------------------------------------------------------------------------
// Shared staging: zero-fill patch, async-copy in-range 17-col span per row.
// patch plane layout: [r 0..2][c 0..17][ch 0..127] halves
// ---------------------------------------------------------------------------
__device__ __forceinline__ void stage_planes(_Float16* patch, int nplanes,
                                             const _Float16* const* srcs,
                                             int lb, int h, int w0, int tid) {
  // zero fill (covers padded borders)
  int nz = nplanes * 3 * 18 * (UU / 2);
  for (int idx = tid; idx < nz; idx += 256) ((unsigned int*)patch)[idx] = 0u;
  __syncthreads();
  // async copies: per plane/row, 17 in-range columns = 272 x 16B
  int c0 = (w0 == 0) ? 1 : 0;
  int ntr = nplanes * 3 * 272;
  for (int i = tid; i < ntr; i += 256) {
    int k  = i % 272;
    int r  = (i / 272) % 3;
    int pl = i / (3 * 272);
    int hh = h + r - 1;
    if (hh < 0 || hh >= HH) continue;
    const _Float16* g = srcs[pl] +
        (((size_t)lb * HH + hh) * WW + (w0 + c0 - 1)) * UU + (size_t)k * 8;
    char* l = (char*)patch + ((((pl * 3 + r) * 18 + c0) * UU) * 2 + k * 16);
    copy16_g2lds(g, l);
  }
  wait_async_copies();
  __syncthreads();
}

// ---------------------------------------------------------------------------
// Kernel 4: conv B as implicit GEMM. block = 16 pixels x 128 channels,
// 8 waves (one 16-channel N tile each), re+im accumulators. grid (LB,H,W/16)
// ---------------------------------------------------------------------------
__global__ __launch_bounds__(256)
void convB_kernel(const _Float16* __restrict__ uh,     // (LB,H,W,U) f16
                  const _Float16* __restrict__ Bw,
                  _Float16* __restrict__ BuRe,
                  _Float16* __restrict__ BuIm) {
  __shared__ __align__(16) _Float16 patch[3 * 18 * UU];
  int lb = blockIdx.x, h = blockIdx.y, w0 = blockIdx.z * 16;
  int tid = threadIdx.x;

  const _Float16* srcs[1] = { uh };
  stage_planes(patch, 1, srcs, lb, h, w0, tid);

  int wave = tid >> 5;
  int lane = tid & 31;
  int m = lane & 15, half = lane >> 4;
  v8f accRe = {}; v8f accIm = {};

  for (int t = 0; t < 9; ++t) {
    int dy = t / 3, dx = t % 3;
    #pragma unroll
    for (int kc = 0; kc < 4; ++kc) {
      const _Float16* rowp = patch + (dy * 18 + (m + dx)) * UU + kc * 32;
      v16h a = make_afrag(rowp, half);
      const unsigned int* base =
          (const unsigned int*)Bw + ((size_t)((t * 4 + kc) * 8 + wave) * 2) * 256;
      v16h bre = load_bfrag(base);
      v16h bim = load_bfrag(base + 256);
      accRe = wmma16x16x32(a, bre, accRe);
      accIm = wmma16x16x32(a, bim, accIm);
    }
  }

  int ch = wave * 16 + (lane & 15);
  #pragma unroll
  for (int r = 0; r < 8; ++r) {
    int M = half * 8 + r;
    size_t o = (((size_t)lb * HH + h) * WW + (w0 + M)) * PP + ch;
    BuRe[o] = (_Float16)accRe[r];
    BuIm[o] = (_Float16)accIm[r];
  }
}

// ---------------------------------------------------------------------------
// Kernel 5: sequential complex scan over L (in place, f16 storage, f32 state).
// ---------------------------------------------------------------------------
__global__ __launch_bounds__(256)
void scan_kernel(const float* __restrict__ ws,
                 const float* __restrict__ x0,
                 _Float16* __restrict__ XRe,
                 _Float16* __restrict__ XIm,
                 float* __restrict__ outLast) {
  size_t idx = (size_t)blockIdx.x * 256 + threadIdx.x;   // over B*H*W*P
  if (idx >= NPIX) return;
  int p = (int)(idx & (PP - 1));
  float Are = ws[p], Aim = ws[128 + p];
  float xr = x0[idx], xi = 0.0f;
  #pragma unroll
  for (int l = 0; l < LSEQ; ++l) {
    size_t o = (size_t)l * NPIX + idx;     // lb = l*B + b
    float br = (float)XRe[o], bi = (float)XIm[o];
    float nr = Are * xr - Aim * xi + br;
    float ni = Are * xi + Aim * xr + bi;
    xr = nr; xi = ni;
    XRe[o] = (_Float16)xr; XIm[o] = (_Float16)xi;
  }
  outLast[2 * idx]     = xr;
  outLast[2 * idx + 1] = xi;
}

// ---------------------------------------------------------------------------
// Kernel 6: conv C (+ depthwise D from LDS u-plane + tanh-GELU).
// K = {re,im} x 9 taps x 128 P.  grid (LB, H, W/16)
// ---------------------------------------------------------------------------
__device__ __forceinline__ float gelu_tanh(float v) {
  float c = 0.7978845608028654f * (v + 0.044715f * v * v * v);
  return 0.5f * v * (1.0f + tanhf(c));
}

__global__ __launch_bounds__(256)
void convC_kernel(const _Float16* __restrict__ XRe,
                  const _Float16* __restrict__ XIm,
                  const _Float16* __restrict__ Cw,
                  const _Float16* __restrict__ uh,
                  const float* __restrict__ Dk,       // (3,3,U)
                  float* __restrict__ ys) {
  __shared__ __align__(16) _Float16 patch[3 * 3 * 18 * PP];  // re, im, u planes
  int lb = blockIdx.x, h = blockIdx.y, w0 = blockIdx.z * 16;
  int tid = threadIdx.x;

  const _Float16* srcs[3] = { XRe, XIm, uh };
  stage_planes(patch, 3, srcs, lb, h, w0, tid);

  int wave = tid >> 5;
  int lane = tid & 31;
  int m = lane & 15, half = lane >> 4;
  v8f acc = {};

  for (int rp = 0; rp < 2; ++rp) {
    for (int t = 0; t < 9; ++t) {
      int dy = t / 3, dx = t % 3;
      #pragma unroll
      for (int kc = 0; kc < 4; ++kc) {
        const _Float16* rowp =
            patch + ((rp * 3 + dy) * 18 + (m + dx)) * PP + kc * 32;
        v16h a = make_afrag(rowp, half);
        const unsigned int* base =
            (const unsigned int*)Cw +
            (size_t)((((rp * 9 + t) * 4 + kc) * 8) + wave) * 256;
        v16h b = load_bfrag(base);
        acc = wmma16x16x32(a, b, acc);
      }
    }
  }

  int ch = wave * 16 + (lane & 15);
  // depthwise D weights for this channel (loop-invariant over rows)
  float dw[9];
  #pragma unroll
  for (int t = 0; t < 9; ++t) dw[t] = Dk[t * UU + ch];

  const _Float16* up = patch + (2 * 3) * 18 * PP;   // u plane
  #pragma unroll
  for (int r = 0; r < 8; ++r) {
    int M = half * 8 + r;
    float du = 0.0f;
    #pragma unroll
    for (int dy = 0; dy < 3; ++dy)
      #pragma unroll
      for (int dx = 0; dx < 3; ++dx)
        du += (float)up[(dy * 18 + (M + dx)) * PP + ch] * dw[dy * 3 + dx];
    float v = gelu_tanh(acc[r] + du);
    ys[(((size_t)lb * HH + h) * WW + (w0 + M)) * UU + ch] = v;
  }
}

// ---------------------------------------------------------------------------
// Launch
// ---------------------------------------------------------------------------
extern "C" void kernel_launch(void* const* d_in, const int* in_sizes, int n_in,
                              void* d_out, int out_size, void* d_ws, size_t ws_size,
                              hipStream_t stream) {
  const float* u    = (const float*)d_in[0];   // (L,B,H,W,U)
  const float* x0   = (const float*)d_in[1];   // (B,H,W,P)
  const float* Lre  = (const float*)d_in[2];
  const float* Lim  = (const float*)d_in[3];
  const float* Bk   = (const float*)d_in[4];   // (3,3,U,P,2)
  const float* Ck   = (const float*)d_in[5];   // (3,3,P,U,2)
  const float* Dk   = (const float*)d_in[6];   // (3,3,U)
  const float* lgs  = (const float*)d_in[7];   // (P,)

  char* wsb = (char*)d_ws;
  float*    wsDisc = (float*)(wsb + O_DISC);
  _Float16* Bw     = (_Float16*)(wsb + O_BW);
  _Float16* Cw     = (_Float16*)(wsb + O_CW);
  _Float16* uh     = (_Float16*)(wsb + O_UH);
  _Float16* BuRe   = (_Float16*)(wsb + O_BURE);
  _Float16* BuIm   = (_Float16*)(wsb + O_BUIM);

  float* out     = (float*)d_out;
  float* outLast = out;                                    // (B,H,W,P) complex
  float* ys      = out + 2 * (size_t)BB * HH * WW * PP;    // (L,B,H,W,U)

  disc_kernel<<<1, 128, 0, stream>>>(Lre, Lim, lgs, wsDisc);
  cvt_kernel<<<(unsigned)(PLANEH / 512), 256, 0, stream>>>(u, uh);
  packB_kernel<<<576, 256, 0, stream>>>(Bk, wsDisc, Bw);
  packC_kernel<<<576, 256, 0, stream>>>(Ck, Cw);

  dim3 cg(LB, HH, WW / 16);
  convB_kernel<<<cg, 256, 0, stream>>>(uh, Bw, BuRe, BuIm);

  scan_kernel<<<(unsigned)(NPIX / 256), 256, 0, stream>>>(
      wsDisc, x0, BuRe, BuIm, outLast);

  convC_kernel<<<cg, 256, 0, stream>>>(BuRe, BuIm, Cw, uh, Dk, ys);
}
